// PointerGeneratorTransformer_31825707664200
// MI455X (gfx1250) — compile-verified, matrix operand order checked by reading, not run on hardware
//
#include <hip/hip_runtime.h>
#include <hip/hip_bf16.h>
#include <math.h>

// Problem constants
#define S_   512
#define N_   16
#define T_   200
#define E_   768
#define V_   21128
#define M_   (T_ * N_)            // 3200 rows of the big GEMM
#define KSTEPS (E_ / 32)          // 24 k-steps of 32
#define MTILES (M_ / 16)          // 200
#define NTILES ((V_ + 15) / 16)   // 1321 (tail tile half-full)
#define NTP    1324               // padded to multiple of 4 (extra tiles all-zero)
#define NQUADS (NTP / 4)          // 331 quads of 16x64 output per wave

typedef __attribute__((ext_vector_type(16))) __bf16 v16bf;
typedef __attribute__((ext_vector_type(8)))  float  v8f;

// ---------------------------------------------------------------------------
// Fragment layout (CDNA5 ISA 7.12.2, 16-bit A 16x32):
//   lanes 0-15  : row = lane,    elems 0..7 -> K 0..7,  elems 8..15 -> K 16..23
//   lanes 16-31 : row = lane-16, elems 0..7 -> K 8..15, elems 8..15 -> K 24..31
// B (32x16) stored symmetrically with N in place of M.
__device__ __forceinline__ int frag_koff(int lane, int e) {
    int half = lane >> 4;
    return (e < 8) ? (e + 8 * half) : (e + 8 + 8 * half);
}

// decoder_output (M_ x E_) fp32 -> split bf16 hi/lo WMMA fragments.
__global__ void prep_a(const float* __restrict__ D,
                       __bf16* __restrict__ Ahi, __bf16* __restrict__ Alo) {
    int idx = blockIdx.x * 256 + threadIdx.x;       // (mtile*KSTEPS + ks)*32 + lane
    if (idx >= MTILES * KSTEPS * 32) return;
    int lane  = idx & 31;
    int mk    = idx >> 5;
    int ks    = mk % KSTEPS;
    int mtile = mk / KSTEPS;
    int m     = mtile * 16 + (lane & 15);
    int kbase = ks * 32;
    size_t fo = (size_t)idx * 16;
    for (int e = 0; e < 16; ++e) {
        float x = D[(size_t)m * E_ + kbase + frag_koff(lane, e)];
        __bf16 h = (__bf16)x;
        __bf16 l = (__bf16)(x - (float)h);
        Ahi[fo + e] = h;
        Alo[fo + e] = l;
    }
}

// W_vocab (E_ x V_) fp32 -> split bf16 hi/lo WMMA fragments (zero beyond V_).
__global__ void prep_b(const float* __restrict__ W,
                       __bf16* __restrict__ Bhi, __bf16* __restrict__ Blo) {
    int idx = blockIdx.x * 256 + threadIdx.x;       // (ntile*KSTEPS + ks)*32 + lane
    if (idx >= NTP * KSTEPS * 32) return;
    int lane  = idx & 31;
    int nk    = idx >> 5;
    int ks    = nk % KSTEPS;
    int ntile = nk / KSTEPS;
    int n     = ntile * 16 + (lane & 15);
    int kbase = ks * 32;
    size_t fo = (size_t)idx * 16;
    for (int e = 0; e < 16; ++e) {
        int k = kbase + frag_koff(lane, e);
        float x = (n < V_) ? W[(size_t)k * V_ + n] : 0.0f;
        __bf16 h = (__bf16)x;
        __bf16 l = (__bf16)(x - (float)h);
        Bhi[fo + e] = h;
        Blo[fo + e] = l;
    }
}

// Split-bf16 WMMA GEMM: Z = D @ W + bias.  One wave -> 16 rows x 64 cols.
// Grid sized exactly: no guards, no divergence; EXEC all-1s through every WMMA.
__global__ void __launch_bounds__(256) gemm_wmma(
        const __bf16* __restrict__ Ahi, const __bf16* __restrict__ Alo,
        const __bf16* __restrict__ Bhi, const __bf16* __restrict__ Blo,
        const float* __restrict__ bias, float* __restrict__ Z) {
    int wid   = blockIdx.x * 8 + (threadIdx.x >> 5);   // uniform per wave
    int lane  = threadIdx.x & 31;
    int mtile = wid / NQUADS;
    int nquad = wid % NQUADS;
    int ntb   = nquad * 4;

    v8f c0 = {0.f,0.f,0.f,0.f,0.f,0.f,0.f,0.f};
    v8f c1 = c0, c2 = c0, c3 = c0;

    const __bf16* pah = Ahi + ((size_t)(mtile * KSTEPS) * 32 + lane) * 16;
    const __bf16* pal = Alo + ((size_t)(mtile * KSTEPS) * 32 + lane) * 16;
    const __bf16* pbh0 = Bhi + ((size_t)((ntb + 0) * KSTEPS) * 32 + lane) * 16;
    const __bf16* pbl0 = Blo + ((size_t)((ntb + 0) * KSTEPS) * 32 + lane) * 16;
    const __bf16* pbh1 = Bhi + ((size_t)((ntb + 1) * KSTEPS) * 32 + lane) * 16;
    const __bf16* pbl1 = Blo + ((size_t)((ntb + 1) * KSTEPS) * 32 + lane) * 16;
    const __bf16* pbh2 = Bhi + ((size_t)((ntb + 2) * KSTEPS) * 32 + lane) * 16;
    const __bf16* pbl2 = Blo + ((size_t)((ntb + 2) * KSTEPS) * 32 + lane) * 16;
    const __bf16* pbh3 = Bhi + ((size_t)((ntb + 3) * KSTEPS) * 32 + lane) * 16;
    const __bf16* pbl3 = Blo + ((size_t)((ntb + 3) * KSTEPS) * 32 + lane) * 16;

    for (int ks = 0; ks < KSTEPS; ++ks) {
        size_t fs = (size_t)ks * 32 * 16;              // per-kstep fragment stride
        v16bf ah = *(const v16bf*)(pah + fs);
        v16bf al = *(const v16bf*)(pal + fs);
        // tile 0:  hi*hi + hi*lo + lo*hi  (fp32 accumulate)
        {
            v16bf bh = *(const v16bf*)(pbh0 + fs);
            v16bf bl = *(const v16bf*)(pbl0 + fs);
            c0 = __builtin_amdgcn_wmma_f32_16x16x32_bf16(false, ah, false, bh, (short)0, c0, false, false);
            c0 = __builtin_amdgcn_wmma_f32_16x16x32_bf16(false, ah, false, bl, (short)0, c0, false, false);
            c0 = __builtin_amdgcn_wmma_f32_16x16x32_bf16(false, al, false, bh, (short)0, c0, false, false);
        }
        {
            v16bf bh = *(const v16bf*)(pbh1 + fs);
            v16bf bl = *(const v16bf*)(pbl1 + fs);
            c1 = __builtin_amdgcn_wmma_f32_16x16x32_bf16(false, ah, false, bh, (short)0, c1, false, false);
            c1 = __builtin_amdgcn_wmma_f32_16x16x32_bf16(false, ah, false, bl, (short)0, c1, false, false);
            c1 = __builtin_amdgcn_wmma_f32_16x16x32_bf16(false, al, false, bh, (short)0, c1, false, false);
        }
        {
            v16bf bh = *(const v16bf*)(pbh2 + fs);
            v16bf bl = *(const v16bf*)(pbl2 + fs);
            c2 = __builtin_amdgcn_wmma_f32_16x16x32_bf16(false, ah, false, bh, (short)0, c2, false, false);
            c2 = __builtin_amdgcn_wmma_f32_16x16x32_bf16(false, ah, false, bl, (short)0, c2, false, false);
            c2 = __builtin_amdgcn_wmma_f32_16x16x32_bf16(false, al, false, bh, (short)0, c2, false, false);
        }
        {
            v16bf bh = *(const v16bf*)(pbh3 + fs);
            v16bf bl = *(const v16bf*)(pbl3 + fs);
            c3 = __builtin_amdgcn_wmma_f32_16x16x32_bf16(false, ah, false, bh, (short)0, c3, false, false);
            c3 = __builtin_amdgcn_wmma_f32_16x16x32_bf16(false, ah, false, bl, (short)0, c3, false, false);
            c3 = __builtin_amdgcn_wmma_f32_16x16x32_bf16(false, al, false, bh, (short)0, c3, false, false);
        }
    }

    // C/D layout: lanes 0-15 -> rows 0..7 in vgprs 0..7; lanes 16-31 -> rows 8..15
    int mbase = mtile * 16 + 8 * (lane >> 4);
    int lcol  = lane & 15;
    #pragma unroll
    for (int i = 0; i < 4; ++i) {
        int col = (ntb + i) * 16 + lcol;
        v8f* c = (i == 0) ? &c0 : (i == 1) ? &c1 : (i == 2) ? &c2 : &c3;
        if (col < V_) {
            float bb = bias[col];
            #pragma unroll
            for (int r = 0; r < 8; ++r)
                Z[(size_t)(mbase + r) * V_ + col] = (*c)[r] + bb;
        }
    }
}

// Per-row softmax stats over V_ (one block per row).
__global__ void rowstats(const float* __restrict__ Z,
                         float* __restrict__ rmax, float* __restrict__ rsum) {
    __shared__ float red[256];
    int m = blockIdx.x;
    int tid = threadIdx.x;
    const float* row = Z + (size_t)m * V_;
    float mx = -INFINITY;
    for (int v = tid; v < V_; v += 256) mx = fmaxf(mx, row[v]);
    red[tid] = mx; __syncthreads();
    for (int s = 128; s > 0; s >>= 1) {
        if (tid < s) red[tid] = fmaxf(red[tid], red[tid + s]);
        __syncthreads();
    }
    float M = red[0]; __syncthreads();
    float sm = 0.f;
    for (int v = tid; v < V_; v += 256) sm += expf(row[v] - M);
    red[tid] = sm; __syncthreads();
    for (int s = 128; s > 0; s >>= 1) {
        if (tid < s) red[tid] += red[tid + s];
        __syncthreads();
    }
    if (tid == 0) { rmax[m] = M; rsum[m] = red[0]; }
}

// context[t,n,:] = sum_s attention[n,t,s] * memory[s,n,:]
__global__ void context_k(const float* __restrict__ attn,
                          const float* __restrict__ mem,
                          float* __restrict__ ctx) {
    int n  = blockIdx.y;
    int t0 = blockIdx.x * 8;
    int e0 = threadIdx.x;
    float acc[8][3];
    #pragma unroll
    for (int j = 0; j < 8; ++j) { acc[j][0] = 0.f; acc[j][1] = 0.f; acc[j][2] = 0.f; }
    for (int s = 0; s < S_; ++s) {
        size_t hb = ((size_t)s * N_ + n) * E_;
        float h0 = mem[hb + e0];
        float h1 = mem[hb + e0 + 256];
        float h2 = mem[hb + e0 + 512];
        #pragma unroll
        for (int j = 0; j < 8; ++j) {
            float a = attn[((size_t)n * T_ + (t0 + j)) * S_ + s];
            acc[j][0] += a * h0; acc[j][1] += a * h1; acc[j][2] += a * h2;
        }
    }
    #pragma unroll
    for (int j = 0; j < 8; ++j) {
        size_t cb = ((size_t)(t0 + j) * N_ + n) * E_;
        ctx[cb + e0]       = acc[j][0];
        ctx[cb + e0 + 256] = acc[j][1];
        ctx[cb + e0 + 512] = acc[j][2];
    }
}

// p_gen[m] = sigmoid([context, dec, emb] . W_gen + b_gen)
__global__ void pgen_k(const float* __restrict__ ctx, const float* __restrict__ dec,
                       const float* __restrict__ emb, const float* __restrict__ Wg,
                       const float* __restrict__ bg, float* __restrict__ pg) {
    __shared__ float red[256];
    int m = blockIdx.x;
    int tid = threadIdx.x;
    size_t rb = (size_t)m * E_;
    float s = 0.f;
    for (int e = tid; e < E_; e += 256) {
        s += ctx[rb + e] * Wg[e]
           + dec[rb + e] * Wg[E_ + e]
           + emb[rb + e] * Wg[2 * E_ + e];
    }
    red[tid] = s; __syncthreads();
    for (int w = 128; w > 0; w >>= 1) {
        if (tid < w) red[tid] += red[tid + w];
        __syncthreads();
    }
    if (tid == 0) pg[m] = 1.f / (1.f + expf(-(red[0] + bg[0])));
}

// out[n,t,v] = p_vocab * p_gen   (initializes every output element)
__global__ void combine1(const float* __restrict__ Z, const float* __restrict__ rmax,
                         const float* __restrict__ rsum, const float* __restrict__ pg,
                         float* __restrict__ out) {
    int m = blockIdx.y;                       // m = t*N_ + n
    int v = blockIdx.x * 256 + threadIdx.x;
    if (v >= V_) return;
    int t = m >> 4, n = m & 15;
    float p = expf(Z[(size_t)m * V_ + v] - rmax[m]) / rsum[m] * pg[m];
    out[((size_t)n * T_ + t) * V_ + v] = p;
}

// out[n,t,src[n,s]] += attn[n,t,s] * (1 - p_gen)
__global__ void scatter_k(const float* __restrict__ attn, const int* __restrict__ src,
                          const float* __restrict__ pg, float* __restrict__ out) {
    int m = blockIdx.x;
    int t = m >> 4, n = m & 15;
    float pc = 1.f - pg[m];
    float* row = out + ((size_t)n * T_ + t) * V_;
    for (int s = threadIdx.x; s < S_; s += 256) {
        float a = attn[((size_t)n * T_ + t) * S_ + s];
        int v = src[n * S_ + s];
        atomicAdd(&row[v], a * pc);
    }
}

__global__ void logk(float* __restrict__ out, size_t total) {
    size_t i = (size_t)blockIdx.x * 256 + threadIdx.x;
    if (i < total) out[i] = logf(out[i]);
}

extern "C" void kernel_launch(void* const* d_in, const int* in_sizes, int n_in,
                              void* d_out, int out_size, void* d_ws, size_t ws_size,
                              hipStream_t stream) {
    const float* memory = (const float*)d_in[0];
    const float* dec    = (const float*)d_in[1];
    const float* emb    = (const float*)d_in[2];
    const float* attn   = (const float*)d_in[3];
    const int*   src    = (const int*)  d_in[4];
    const float* Wv     = (const float*)d_in[5];
    const float* bv     = (const float*)d_in[6];
    const float* Wg     = (const float*)d_in[7];
    const float* bg     = (const float*)d_in[8];
    float* out = (float*)d_out;
    char*  ws  = (char*)d_ws;

    size_t off = 0;
    auto alloc = [&](size_t bytes) { size_t o = off; off = (off + bytes + 255) & ~(size_t)255; return o; };
    size_t afrag = (size_t)MTILES * KSTEPS * 32 * 16;   // bf16 elems
    size_t bfrag = (size_t)NTP * KSTEPS * 32 * 16;
    float*  Z    = (float*)(ws + alloc((size_t)M_ * V_ * sizeof(float)));
    __bf16* Ahi  = (__bf16*)(ws + alloc(afrag * sizeof(__bf16)));
    __bf16* Alo  = (__bf16*)(ws + alloc(afrag * sizeof(__bf16)));
    __bf16* Bhi  = (__bf16*)(ws + alloc(bfrag * sizeof(__bf16)));
    __bf16* Blo  = (__bf16*)(ws + alloc(bfrag * sizeof(__bf16)));
    float*  ctx  = (float*)(ws + alloc((size_t)M_ * E_ * sizeof(float)));
    float*  pg   = (float*)(ws + alloc((size_t)M_ * sizeof(float)));
    float*  rmax = (float*)(ws + alloc((size_t)M_ * sizeof(float)));
    float*  rsum = (float*)(ws + alloc((size_t)M_ * sizeof(float)));

    // 1) fragment prep
    {
        int tot = MTILES * KSTEPS * 32;
        prep_a<<<(tot + 255) / 256, 256, 0, stream>>>(dec, Ahi, Alo);
    }
    {
        int tot = NTP * KSTEPS * 32;
        prep_b<<<(tot + 255) / 256, 256, 0, stream>>>(Wv, Bhi, Blo);
    }
    // 2) big GEMM via WMMA: one wave per 16x64 slab; grid exactly covers work
    {
        int waves = MTILES * NQUADS;                 // 66,200 -> 8,275 blocks
        gemm_wmma<<<waves / 8, 256, 0, stream>>>(Ahi, Alo, Bhi, Blo, bv, Z);
    }
    // 3) softmax row stats
    rowstats<<<M_, 256, 0, stream>>>(Z, rmax, rsum);
    // 4) context vectors
    {
        dim3 g(T_ / 8, N_);
        context_k<<<g, 256, 0, stream>>>(attn, memory, ctx);
    }
    // 5) p_gen
    pgen_k<<<M_, 256, 0, stream>>>(ctx, dec, emb, Wg, bg, pg);
    // 6) out = p_vocab * p_gen
    {
        dim3 g((V_ + 255) / 256, M_);
        combine1<<<g, 256, 0, stream>>>(Z, rmax, rsum, pg, out);
    }
    // 7) scatter copy-distribution with fp32 global atomics
    scatter_k<<<M_, 256, 0, stream>>>(attn, src, pg, out);
    // 8) log in place
    {
        size_t total = (size_t)M_ * V_;
        logk<<<(unsigned)((total + 255) / 256), 256, 0, stream>>>(out, total);
    }
}
